// PairEmbed_34866544509088
// MI455X (gfx1250) — matrix operand order repaired and checked by reading, not compiled, not run on hardware
//
#include <hip/hip_runtime.h>
#include <math.h>

#define EPSF     1e-8f
#define BN_EPSF  1e-3f
#define NPART    128
#define NBATCH   128
#define NPAIR    8256          // 128*129/2
#define NTILES   65            // ceil(8256 / 128)
#define PI_F     3.14159265358979f
#define TWO_PI_F 6.28318530717959f

typedef __attribute__((ext_vector_type(16))) _Float16 v16h;
typedef __attribute__((ext_vector_type(8)))  _Float16 v8h;
typedef __attribute__((ext_vector_type(8)))  float    v8f;

// per-wave LDS ordering fence: DS ops are in-order per wave on CDNA5;
// the asm memory clobber also stops compile-time reordering.
#define LDS_FENCE() asm volatile("s_wait_dscnt 0" ::: "memory")

struct PtrPack { const float* p[22]; };
// p[0..3]  : bn_in  gamma,beta,mean,var            (4 each)
// p[4..9]  : layer1 W(4x64),b,g,beta,mean,var
// p[10..15]: layer2 W(64x64),b,g,beta,mean,var
// p[16..21]: layer3 W(64x8),b,g,beta,mean,var

// ---------------- prep 1: per-particle kinematics ----------------
__global__ void prep_particles(const float* __restrict__ x, float* __restrict__ part) {
    int idx = blockIdx.x * blockDim.x + threadIdx.x;   // b*128+s
    if (idx >= NBATCH * NPART) return;
    int b = idx >> 7, s = idx & 127;
    const float* xb = x + (size_t)b * 4 * NPART + s;
    float px = xb[0], py = xb[NPART], pz = xb[2 * NPART], E = xb[3 * NPART];
    float pt  = sqrtf(fmaxf(px * px + py * py, EPSF));
    float rap = 0.5f * log1pf(2.0f * pz / fmaxf(E - pz, 1e-20f));
    float phi = atan2f(py, px);
    float* o = part + (size_t)idx * 8;
    o[0] = px; o[1] = py; o[2] = pz; o[3] = E;
    o[4] = pt; o[5] = rap; o[6] = phi; o[7] = 0.0f;
}

// ---------------- prep 2: BN-folded f16 weights + tril table ----------------
__global__ void prep_weights(PtrPack P,
                             _Float16* __restrict__ wt1,  // [64][32] d-major, K padded 4->32
                             _Float16* __restrict__ wt2,  // [64][64] d-major
                             _Float16* __restrict__ wt3,  // [16][64] d-major, D padded 8->16
                             float* __restrict__ bias,    // [0..63]=b1 [64..127]=b2 [128..143]=b3
                             int2* __restrict__ ij) {
    const int tid = threadIdx.x;
    const float *g0 = P.p[0], *be0 = P.p[1], *mu0 = P.p[2], *v0 = P.p[3];
    const float *W1 = P.p[4],  *b1 = P.p[5],  *g1 = P.p[6],  *be1 = P.p[7],  *mu1 = P.p[8],  *v1 = P.p[9];
    const float *W2 = P.p[10], *b2 = P.p[11], *g2 = P.p[12], *be2 = P.p[13], *mu2 = P.p[14], *v2 = P.p[15];
    const float *W3 = P.p[16], *b3 = P.p[17], *g3 = P.p[18], *be3 = P.p[19], *mu3 = P.p[20], *v3 = P.p[21];

    for (int t = tid; t < 64 * 32; t += 256) {           // layer1 weights (bn_in folded in)
        int d = t >> 5, k = t & 31;
        float s1 = g1[d] * rsqrtf(v1[d] + BN_EPSF);
        float val = 0.0f;
        if (k < 4) {
            float s0 = g0[k] * rsqrtf(v0[k] + BN_EPSF);
            val = W1[k * 64 + d] * s0 * s1;
        }
        wt1[t] = (_Float16)val;
    }
    for (int d = tid; d < 64; d += 256) {                // layer1 bias
        float s1 = g1[d] * rsqrtf(v1[d] + BN_EPSF);
        float t1 = be1[d] - mu1[d] * s1;
        float acc = b1[d];
        for (int c = 0; c < 4; ++c) {
            float s0 = g0[c] * rsqrtf(v0[c] + BN_EPSF);
            float t0 = be0[c] - mu0[c] * s0;
            acc += W1[c * 64 + d] * t0;
        }
        bias[d] = s1 * acc + t1;
    }
    for (int t = tid; t < 64 * 64; t += 256) {           // layer2
        int d = t >> 6, k = t & 63;
        float s2 = g2[d] * rsqrtf(v2[d] + BN_EPSF);
        wt2[t] = (_Float16)(W2[k * 64 + d] * s2);
    }
    for (int d = tid; d < 64; d += 256) {
        float s2 = g2[d] * rsqrtf(v2[d] + BN_EPSF);
        bias[64 + d] = s2 * b2[d] + (be2[d] - mu2[d] * s2);
    }
    for (int t = tid; t < 16 * 64; t += 256) {           // layer3 (D padded to 16)
        int d = t >> 6, k = t & 63;
        float val = 0.0f;
        if (d < 8) {
            float s3 = g3[d] * rsqrtf(v3[d] + BN_EPSF);
            val = W3[k * 8 + d] * s3;
        }
        wt3[t] = (_Float16)val;
    }
    for (int d = tid; d < 16; d += 256) {
        float val = 0.0f;
        if (d < 8) {
            float s3 = g3[d] * rsqrtf(v3[d] + BN_EPSF);
            val = s3 * b3[d] + (be3[d] - mu3[d] * s3);
        }
        bias[128 + d] = val;
    }
    for (int p = tid; p < NPAIR; p += 256) {             // tril (i,j) table
        int i = (int)((sqrtf(8.0f * (float)p + 1.0f) - 1.0f) * 0.5f);
        while ((i + 1) * (i + 2) / 2 <= p) ++i;
        while (i * (i + 1) / 2 > p) --i;
        ij[p] = make_int2(i, p - i * (i + 1) / 2);
    }
}

// Branch-free exact-GELU: erf via Abramowitz-Stegun 7.1.26 (|err| <= 1.5e-7,
// well below the f16 operand quantization). One native rcp + one native exp,
// six FMAs, no exec divergence.
__device__ __forceinline__ float gelu_fast(float x) {
    float ax = fabsf(x) * 0.70710678118654752f;          // |x|/sqrt(2)
    float t  = __builtin_amdgcn_rcpf(fmaf(0.3275911f, ax, 1.0f));
    float p  = t * fmaf(t, fmaf(t, fmaf(t, fmaf(t, 1.061405429f, -1.453152027f),
                                        1.421413741f), -0.284496736f), 0.254829592f);
    float e    = __expf(-ax * ax);                       // native v_exp_f32
    float erfv = copysignf(fmaf(-p, e, 1.0f), x);        // erf(x/sqrt(2))
    return 0.5f * x * (1.0f + erfv);
}

__device__ __forceinline__ v16h cat16(v8h lo, v8h hi) {
    return __builtin_shufflevector(lo, hi, 0, 1, 2, 3, 4, 5, 6, 7, 8, 9, 10, 11, 12, 13, 14, 15);
}

// ---------------- main fused kernel ----------------
__global__ __launch_bounds__(256) void pair_mlp(const float* __restrict__ part,
                                                const _Float16* __restrict__ wt1g,
                                                const _Float16* __restrict__ wt2g,
                                                const _Float16* __restrict__ wt3g,
                                                const float* __restrict__ biasg,
                                                const int2* __restrict__ ij,
                                                float* __restrict__ out) {
    __shared__ _Float16 lw1[64 * 32];
    __shared__ _Float16 lw2[64 * 64];
    __shared__ _Float16 lw3[16 * 64];
    __shared__ float    lbias[144];
    __shared__ _Float16 act[8][16][72];   // per-wave 16 pairs x 64 ch (stride 72 vs bank conflicts)
    __shared__ float    o3[8][16][8];

    const int tid = threadIdx.x;
    const int b   = blockIdx.y;

    // stage folded weights once per block
    ((uint4*)lw1)[tid]       = ((const uint4*)wt1g)[tid];
    ((uint4*)lw2)[tid]       = ((const uint4*)wt2g)[tid];
    ((uint4*)lw2)[tid + 256] = ((const uint4*)wt2g)[tid + 256];
    if (tid < 128) ((uint4*)lw3)[tid] = ((const uint4*)wt3g)[tid];
    if (tid < 144) lbias[tid] = biasg[tid];
    __syncthreads();

    const int wave  = tid >> 5;
    const int lane  = tid & 31;
    const int m     = lane & 15;          // row within 16-pair tile / column within N-tile
    const int khalf = (lane >> 4) & 1;    // which half of the wave
    const int p0    = (blockIdx.x * 8 + wave) * 16;

    // ---- pair features (lanes 0-15 compute; lanes 16-31 zero the K-pad) ----
    if (lane < 16) {
        int p  = p0 + m;
        int pc = p < NPAIR ? p : NPAIR - 1;
        int2 pr = ij[pc];
        const float* pi = part + ((size_t)b * NPART + pr.x) * 8;
        const float* pj = part + ((size_t)b * NPART + pr.y) * 8;
        float pxi = pi[0], pyi = pi[1], pzi = pi[2], Ei = pi[3], pti = pi[4], rapi = pi[5], phii = pi[6];
        float pxj = pj[0], pyj = pj[1], pzj = pj[2], Ej = pj[3], ptj = pj[4], rapj = pj[5], phij = pj[6];
        float dphi = fmodf(phii - phij + PI_F, TWO_PI_F);
        if (dphi < 0.0f) dphi += TWO_PI_F;
        dphi -= PI_F;
        float drap    = rapi - rapj;
        float delta   = sqrtf(drap * drap + dphi * dphi);
        float lndelta = logf(fmaxf(delta, EPSF));
        float ptmin   = fminf(pti, ptj);
        float lnkt    = logf(fmaxf(ptmin * delta, EPSF));
        float lnz     = logf(fmaxf(ptmin / fmaxf(pti + ptj, EPSF), EPSF));
        float Es = Ei + Ej, pxs = pxi + pxj, pys = pyi + pyj, pzs = pzi + pzj;
        float m2   = Es * Es - (pxs * pxs + pys * pys + pzs * pzs);
        float lnm2 = logf(fmaxf(m2, EPSF));
        v8h f = {(_Float16)lnkt, (_Float16)lnz, (_Float16)lndelta, (_Float16)lnm2,
                 (_Float16)0.0f, (_Float16)0.0f, (_Float16)0.0f, (_Float16)0.0f};
        *(v8h*)&act[wave][m][0] = f;
    } else {
        v8h z = {};
        *(v8h*)&act[wave][m][8]  = z;
        *(v8h*)&act[wave][m][16] = z;
        *(v8h*)&act[wave][m][24] = z;
    }
    LDS_FENCE();

    const int kb = khalf * 16;            // A-fragment K base per lane half

    // ---- layer 1: 4(pad32) -> 64, one WMMA per 16-col N-tile ----
    {
        v16h a1 = cat16(*(const v8h*)&act[wave][m][kb], *(const v8h*)&act[wave][m][kb + 8]);
        #pragma unroll
        for (int n = 0; n < 4; ++n) {
            int d = n * 16 + m;           // output channel owned by this lane
            v16h bf = cat16(*(const v8h*)&lw1[d * 32 + khalf * 16],
                            *(const v8h*)&lw1[d * 32 + khalf * 16 + 8]);
            float bs = lbias[d];
            v8f c = {bs, bs, bs, bs, bs, bs, bs, bs};
            c = __builtin_amdgcn_wmma_f32_16x16x32_f16(false, a1, false, bf, (short)0, c, false, false);
            #pragma unroll
            for (int r = 0; r < 8; ++r)
                act[wave][khalf * 8 + r][d] = (_Float16)gelu_fast(c[r]);
        }
    }
    LDS_FENCE();

    // ---- layer 2: 64 -> 64, two WMMAs (K=64) per N-tile ----
    {
        v16h a0 = cat16(*(const v8h*)&act[wave][m][kb],      *(const v8h*)&act[wave][m][kb + 8]);
        v16h a1 = cat16(*(const v8h*)&act[wave][m][32 + kb], *(const v8h*)&act[wave][m][32 + kb + 8]);
        #pragma unroll
        for (int n = 0; n < 4; ++n) {
            int d = n * 16 + m;
            v16h b0 = cat16(*(const v8h*)&lw2[d * 64 + khalf * 16],
                            *(const v8h*)&lw2[d * 64 + khalf * 16 + 8]);
            v16h b1 = cat16(*(const v8h*)&lw2[d * 64 + 32 + khalf * 16],
                            *(const v8h*)&lw2[d * 64 + 32 + khalf * 16 + 8]);
            float bs = lbias[64 + d];
            v8f c = {bs, bs, bs, bs, bs, bs, bs, bs};
            c = __builtin_amdgcn_wmma_f32_16x16x32_f16(false, a0, false, b0, (short)0, c, false, false);
            c = __builtin_amdgcn_wmma_f32_16x16x32_f16(false, a1, false, b1, (short)0, c, false, false);
            #pragma unroll
            for (int r = 0; r < 8; ++r)
                act[wave][khalf * 8 + r][d] = (_Float16)gelu_fast(c[r]);
        }
    }
    LDS_FENCE();

    // ---- layer 3: 64 -> 8 (padded to 16), single N-tile ----
    {
        v16h a0 = cat16(*(const v8h*)&act[wave][m][kb],      *(const v8h*)&act[wave][m][kb + 8]);
        v16h a1 = cat16(*(const v8h*)&act[wave][m][32 + kb], *(const v8h*)&act[wave][m][32 + kb + 8]);
        v16h b0 = cat16(*(const v8h*)&lw3[m * 64 + khalf * 16],
                        *(const v8h*)&lw3[m * 64 + khalf * 16 + 8]);
        v16h b1 = cat16(*(const v8h*)&lw3[m * 64 + 32 + khalf * 16],
                        *(const v8h*)&lw3[m * 64 + 32 + khalf * 16 + 8]);
        float bs = lbias[128 + m];
        v8f c = {bs, bs, bs, bs, bs, bs, bs, bs};
        c = __builtin_amdgcn_wmma_f32_16x16x32_f16(false, a0, false, b0, (short)0, c, false, false);
        c = __builtin_amdgcn_wmma_f32_16x16x32_f16(false, a1, false, b1, (short)0, c, false, false);
        if (m < 8) {
            #pragma unroll
            for (int r = 0; r < 8; ++r) o3[wave][khalf * 8 + r][m] = c[r];
        }
    }
    LDS_FENCE();

    // ---- symmetric scatter into y(B,8,S,S); consecutive pairs -> consecutive j ----
    #pragma unroll
    for (int t = 0; t < 4; ++t) {
        int c8 = t * 2 + khalf;
        float v = o3[wave][m][c8];
        int p = p0 + m;
        if (p < NPAIR) {
            int2 pr = ij[p];
            size_t base = ((size_t)b * 8 + c8) * NPART;
            out[(base + pr.x) * NPART + pr.y] = v;
            out[(base + pr.y) * NPART + pr.x] = v;
        }
    }
}

extern "C" void kernel_launch(void* const* d_in, const int* in_sizes, int n_in,
                              void* d_out, int out_size, void* d_ws, size_t ws_size,
                              hipStream_t stream) {
    // input order: x first (insertion order) or params first (sorted pytree) — detect via size
    int xi, base;
    if (in_sizes[0] == NBATCH * 4 * NPART) { xi = 0; base = 1; }
    else                                   { base = 0; xi = n_in - 1; }
    const float* x = (const float*)d_in[xi];
    PtrPack pk;
    for (int t = 0; t < 22; ++t) pk.p[t] = (const float*)d_in[base + t];

    char* ws = (char*)d_ws;
    float*     part = (float*)(ws);                 // 128*128*8*4 = 524288 B
    _Float16*  wt1  = (_Float16*)(ws + 524288);     // 4096 B
    _Float16*  wt2  = (_Float16*)(ws + 528384);     // 8192 B
    _Float16*  wt3  = (_Float16*)(ws + 536576);     // 2048 B
    float*     bias = (float*)(ws + 538624);        // 576 B
    int2*      ij   = (int2*)(ws + 539264);         // 66048 B
    float*     out  = (float*)d_out;

    prep_particles<<<dim3((NBATCH * NPART + 255) / 256), 256, 0, stream>>>(x, part);
    prep_weights<<<dim3(1), 256, 0, stream>>>(pk, wt1, wt2, wt3, bias, ij);
    pair_mlp<<<dim3(NTILES, NBATCH), 256, 0, stream>>>(part, wt1, wt2, wt3, bias, ij, out);
}
